// GCN_57621281243368
// MI455X (gfx1250) — matrix operand order repaired and checked by reading, not compiled, not run on hardware
//
#include <hip/hip_runtime.h>
#include <math.h>
#include <stdint.h>

typedef __attribute__((ext_vector_type(16))) _Float16 v16h;
typedef __attribute__((ext_vector_type(8)))  _Float16 v8h;
typedef __attribute__((ext_vector_type(4)))  _Float16 v4h;
typedef __attribute__((ext_vector_type(8)))  float    v8f;
typedef __attribute__((ext_vector_type(4)))  float    v4f;

#define DIN    512
#define DH     256
#define DOUT   47
#define DOUT_P 48

// ---------------------------------------------------------------------------
// f32 -> f16 row-major conversion, 4 elements per thread (vectorized)
// ---------------------------------------------------------------------------
__global__ void cvt_f32_f16_v4(const float* __restrict__ in,
                               _Float16* __restrict__ out, int n4) {
    int t = blockIdx.x * blockDim.x + threadIdx.x;
    if (t >= n4) return;
    v4f v = *(const v4f*)(in + (size_t)t * 4);
    v4h h;
    h[0] = (_Float16)v[0]; h[1] = (_Float16)v[1];
    h[2] = (_Float16)v[2]; h[3] = (_Float16)v[3];
    *(v4h*)(out + (size_t)t * 4) = h;
}

// ---------------------------------------------------------------------------
// Pack a KxN f32 weight matrix into WMMA B-fragment order (f16).
// Fragment f = kt*n_tiles + nt holds 512 halves: [lane*16 + j] where
//   K-row = kt*32 + 16*(lane>=16) + j ,  col = nt*16 + (lane&15).
// Columns >= N are zero-padded.
// ---------------------------------------------------------------------------
__global__ void pack_b_frags(const float* __restrict__ W,
                             _Float16* __restrict__ out,
                             int K, int N, int n_tiles) {
    int t = blockIdx.x * blockDim.x + threadIdx.x;
    int total = (K >> 5) * n_tiles * 512;
    if (t >= total) return;
    int f = t >> 9, r = t & 511;
    int l = r >> 4, j = r & 15;
    int kt = f / n_tiles, nt = f % n_tiles;
    int krow = kt * 32 + ((l >> 4) << 4) + j;
    int col  = nt * 16 + (l & 15);
    float v = (col < N) ? W[(size_t)krow * N + col] : 0.0f;
    out[t] = (_Float16)v;
}

// ---------------------------------------------------------------------------
// LDS-staged WMMA GEMM: C[M x NSTRIDE] = A[M x K](f16 row-major) * Bp(frags).
// A block owns STRIPES 16-row stripes of A, staged once into LDS via
// global_load_async_to_lds_b128 (ASYNCcnt) and read back via ds_load_b128.
// Each wave computes NTILE adjacent 16x16 tiles, reusing its A fragment.
//   layer1: <512,256,2,1>  8 waves cover all 16 column tiles of one stripe
//   layer2: <256, 48,3,4>  4 waves, each covering a full 48-col stripe
// ---------------------------------------------------------------------------
template<int K, int NSTRIDE, int NTILE, int STRIPES>
__global__ void wmma_gemm_lds(const _Float16* __restrict__ A,
                              const _Float16* __restrict__ Bp,
                              float* __restrict__ C, int M) {
    constexpr int NT_TOTAL  = NSTRIDE / 16;
    constexpr int NT_GROUPS = NT_TOTAL / NTILE;       // waves per stripe
    constexpr int NWAVES    = STRIPES * NT_GROUPS;    // waves per block
    constexpr int KP        = K + 8;                  // padded LDS row (halves)
    constexpr int ROWS      = STRIPES * 16;
    constexpr int CHUNKS    = ROWS * (K / 8);         // 16-byte chunks

    __shared__ _Float16 smem[ROWS * KP];

    const int tid  = threadIdx.x;
    const int lane = tid & 31;
    const int wid  = tid >> 5;
    const int row0 = blockIdx.x * ROWS;

    // ---- cooperative async stage: global A stripe -> LDS ----
    for (int c = tid; c < CHUNKS; c += NWAVES * 32) {
        int r  = c / (K / 8);
        int kc = (c % (K / 8)) * 8;
        int gr = row0 + r; if (gr >= M) gr = M - 1;   // clamp tail reads
        unsigned long long gaddr =
            (unsigned long long)(uintptr_t)(A + (size_t)gr * K + kc);
        unsigned laddr = (unsigned)(uintptr_t)(smem + r * KP + kc);
        asm volatile("global_load_async_to_lds_b128 %0, %1, off"
                     :: "v"(laddr), "v"(gaddr) : "memory");
    }
    asm volatile("s_wait_asynccnt 0x0" ::: "memory");
    __syncthreads();

    const int s      = wid / NT_GROUPS;
    const int g      = wid % NT_GROUPS;
    const int lane15 = lane & 31 & 15;
    const int kbase  = (lane >> 4) * 16;              // lanes 16-31 start at K=16

    const _Float16* arow = smem + (s * 16 + lane15) * KP + kbase;
    const _Float16* bp   = Bp + (size_t)lane * 16;

    v8f acc[NTILE];
    #pragma unroll
    for (int t = 0; t < NTILE; ++t) acc[t] = (v8f){};

    #pragma unroll 4
    for (int k0 = 0; k0 < K; k0 += 32) {
        v8h lo = *(const v8h*)(arow + k0);            // K = kbase+k0 .. +7
        v8h hi = *(const v8h*)(arow + k0 + 8);        // K = kbase+k0+8 .. +15
        v16h a = __builtin_shufflevector(lo, hi,
                 0,1,2,3,4,5,6,7,8,9,10,11,12,13,14,15);
        #pragma unroll
        for (int t = 0; t < NTILE; ++t) {
            v16h b = *(const v16h*)(bp +
                     ((size_t)(k0 >> 5) * NT_TOTAL + (g * NTILE + t)) * 512);
            acc[t] = __builtin_amdgcn_wmma_f32_16x16x32_f16(
                         false, a, false, b, (short)0, acc[t], false, false);
        }
    }

    const int mrow0 = row0 + s * 16 + (lane >> 4) * 8;
    #pragma unroll
    for (int t = 0; t < NTILE; ++t) {
        int col = (g * NTILE + t) * 16 + lane15;
        #pragma unroll
        for (int i = 0; i < 8; ++i) {
            int m = mrow0 + i;
            if (m < M) C[(size_t)m * NSTRIDE + col] = acc[t][i];
        }
    }
}

// ---------------------------------------------------------------------------
// Degree handling: deg = 1 (self loop) + scatter-count, then rsqrt.
// ---------------------------------------------------------------------------
__global__ void deg_init(float* __restrict__ dis, int n) {
    int t = blockIdx.x * blockDim.x + threadIdx.x;
    if (t < n) dis[t] = 1.0f;
}
__global__ void deg_accum(const int* __restrict__ dst, float* __restrict__ dis, int E) {
    int t = blockIdx.x * blockDim.x + threadIdx.x;
    if (t < E) atomicAdd(&dis[dst[t]], 1.0f);
}
__global__ void deg_fin(float* __restrict__ dis, int n) {
    int t = blockIdx.x * blockDim.x + threadIdx.x;
    if (t < n) dis[t] = rsqrtf(dis[t]);
}

// ---------------------------------------------------------------------------
// Edge aggregation, 256 features: one wave per edge (incl. self loops);
// b128 gather of h[src], coalesced f32 atomic scatter-add into out[dst].
// ---------------------------------------------------------------------------
__global__ void aggregate256(const float* __restrict__ h,
                             const int* __restrict__ src,
                             const int* __restrict__ dst,
                             const float* __restrict__ dis,
                             float* __restrict__ out, int E, int n_nodes) {
    int wave = (blockIdx.x * blockDim.x + threadIdx.x) >> 5;
    int lane = threadIdx.x & 31;
    if (wave >= E + n_nodes) return;
    int s, d;
    if (wave < E) { s = src[wave]; d = dst[wave]; }
    else          { s = d = wave - E; }               // self loop
    float nrm = dis[s] * dis[d];
    const float* hp = h + (size_t)s * DH + lane * 4;
    v4f va = *(const v4f*)(hp);
    v4f vb = *(const v4f*)(hp + 128);
    float* op = out + (size_t)d * DH + lane * 4;
    #pragma unroll
    for (int i = 0; i < 4; ++i) atomicAdd(&op[i],       va[i] * nrm);
    #pragma unroll
    for (int i = 0; i < 4; ++i) atomicAdd(&op[128 + i], vb[i] * nrm);
}

// ---------------------------------------------------------------------------
// Edge aggregation, 47 features (input stride 48, output stride 47).
// ---------------------------------------------------------------------------
__global__ void aggregate47(const float* __restrict__ h,
                            const int* __restrict__ src,
                            const int* __restrict__ dst,
                            const float* __restrict__ dis,
                            float* __restrict__ out, int E, int n_nodes) {
    int wave = (blockIdx.x * blockDim.x + threadIdx.x) >> 5;
    int lane = threadIdx.x & 31;
    if (wave >= E + n_nodes) return;
    int s, d;
    if (wave < E) { s = src[wave]; d = dst[wave]; }
    else          { s = d = wave - E; }
    float nrm = dis[s] * dis[d];
    const float* hp = h + (size_t)s * DOUT_P;
    float* op = out + (size_t)d * DOUT;
    if (lane < DOUT)      atomicAdd(&op[lane],      hp[lane]      * nrm);
    if (lane + 32 < DOUT) atomicAdd(&op[lane + 32], hp[lane + 32] * nrm);
}

// ---------------------------------------------------------------------------
// a = relu(a + b1), also emit f16 copy for the layer-2 WMMA GEMM.
// ---------------------------------------------------------------------------
__global__ void bias_relu_h16(float* __restrict__ a, const float* __restrict__ bias,
                              _Float16* __restrict__ ah, int n_nodes) {
    int t = blockIdx.x * blockDim.x + threadIdx.x;
    if (t >= n_nodes * (DH / 4)) return;
    int node = t / (DH / 4);
    int f4   = (t % (DH / 4)) * 4;
    float* p = a + (size_t)node * DH + f4;
    v4f v = *(v4f*)p;
    v4h h;
    #pragma unroll
    for (int i = 0; i < 4; ++i) {
        v[i] = fmaxf(v[i] + bias[f4 + i], 0.0f);
        h[i] = (_Float16)v[i];
    }
    *(v4f*)p = v;
    *(v4h*)(ah + (size_t)node * DH + f4) = h;
}

// ---------------------------------------------------------------------------
// out = log_softmax(out + b2) per row of 47; one wave per node.
// ---------------------------------------------------------------------------
__global__ void log_softmax_bias(float* __restrict__ out,
                                 const float* __restrict__ bias, int n_nodes) {
    int wave = (blockIdx.x * blockDim.x + threadIdx.x) >> 5;
    int lane = threadIdx.x & 31;
    if (wave >= n_nodes) return;
    float* row = out + (size_t)wave * DOUT;
    float v0 = (lane      < DOUT) ? row[lane]      + bias[lane]      : -INFINITY;
    float v1 = (lane + 32 < DOUT) ? row[lane + 32] + bias[lane + 32] : -INFINITY;
    float m = fmaxf(v0, v1);
    for (int o = 16; o; o >>= 1) m = fmaxf(m, __shfl_xor(m, o, 32));
    float s = expf(v0 - m) + expf(v1 - m);
    for (int o = 16; o; o >>= 1) s += __shfl_xor(s, o, 32);
    float l = m + logf(s);
    if (lane      < DOUT) row[lane]      = v0 - l;
    if (lane + 32 < DOUT) row[lane + 32] = v1 - l;
}

// ---------------------------------------------------------------------------
extern "C" void kernel_launch(void* const* d_in, const int* in_sizes, int n_in,
                              void* d_out, int out_size, void* d_ws, size_t ws_size,
                              hipStream_t stream) {
    const float* x  = (const float*)d_in[0];
    const int*   ei = (const int*)  d_in[1];
    const float* W1 = (const float*)d_in[2];
    const float* b1 = (const float*)d_in[3];
    const float* W2 = (const float*)d_in[4];
    const float* b2 = (const float*)d_in[5];

    const int N = in_sizes[0] / DIN;     // 40000 nodes
    const int E = in_sizes[1] / 2;       // 640000 edges
    const int* e_src = ei;
    const int* e_dst = ei + E;

    // --- workspace carve (256B aligned) ---
    char* ws = (char*)d_ws;
    auto carve = [&](size_t bytes) {
        char* p = ws;
        ws += (bytes + 255) & ~(size_t)255;
        return p;
    };
    float*    dis = (float*)   carve((size_t)N * sizeof(float));
    float*    h1  = (float*)   carve((size_t)N * DH * sizeof(float));
    float*    a1  = (float*)   carve((size_t)N * DH * sizeof(float));
    float*    h2  = (float*)   carve((size_t)N * DOUT_P * sizeof(float));
    _Float16* xh  = (_Float16*)carve((size_t)N * DIN * sizeof(_Float16));
    _Float16* a1h = (_Float16*)carve((size_t)N * DH * sizeof(_Float16));
    _Float16* w1h = (_Float16*)carve((size_t)(DIN / 32) * (DH / 16) * 512 * sizeof(_Float16));
    _Float16* w2h = (_Float16*)carve((size_t)(DH / 32) * (DOUT_P / 16) * 512 * sizeof(_Float16));

    const int B = 256;
    auto blocks = [](long long work, int per) { return (int)((work + per - 1) / per); };

    // zero accumulation targets (graph-capturable async memsets)
    hipMemsetAsync(a1, 0, (size_t)N * DH * sizeof(float), stream);
    hipMemsetAsync(d_out, 0, (size_t)out_size * sizeof(float), stream);

    // degrees -> deg_inv_sqrt
    deg_init <<<blocks(N, B), B, 0, stream>>>(dis, N);
    deg_accum<<<blocks(E, B), B, 0, stream>>>(e_dst, dis, E);
    deg_fin  <<<blocks(N, B), B, 0, stream>>>(dis, N);

    // operand prep for WMMA
    cvt_f32_f16_v4<<<blocks((long long)N * DIN / 4, B), B, 0, stream>>>(x, xh, N * DIN / 4);
    pack_b_frags<<<blocks((DIN / 32) * (DH / 16) * 512, B), B, 0, stream>>>(W1, w1h, DIN, DH, DH / 16);
    pack_b_frags<<<blocks((DH / 32) * (DOUT_P / 16) * 512, B), B, 0, stream>>>(W2, w2h, DH, DOUT, DOUT_P / 16);

    // layer 1: h1 = x @ W1 (LDS-staged WMMA); aggregate; relu(+b1); f16 recast
    {
        int m_tiles = (N + 15) / 16;                      // 2500
        wmma_gemm_lds<DIN, DH, 2, 1><<<m_tiles, 256, 0, stream>>>(xh, w1h, h1, N);
    }
    aggregate256<<<blocks((long long)(E + N) * 32, B), B, 0, stream>>>(h1, e_src, e_dst,
                                                                       dis, a1, E, N);
    bias_relu_h16<<<blocks((long long)N * (DH / 4), B), B, 0, stream>>>(a1, b1, a1h, N);

    // layer 2: h2 = relu(a1) @ W2 ; aggregate into d_out ; log_softmax(+b2)
    {
        int m_tiles = (N + 15) / 16;
        int grid    = (m_tiles + 3) / 4;                  // 4 stripes per block
        wmma_gemm_lds<DH, DOUT_P, 3, 4><<<grid, 128, 0, stream>>>(a1h, w2h, h2, N);
    }
    aggregate47<<<blocks((long long)(E + N) * 32, B), B, 0, stream>>>(h2, e_src, e_dst, dis,
                                                                      (float*)d_out, E, N);
    log_softmax_bias<<<blocks((long long)N * 32, B), B, 0, stream>>>((float*)d_out, b2, N);
}
